// TextDecoder_9474697855669
// MI455X (gfx1250) — compile-verified, matrix operand order checked by reading, not run on hardware
//
#include <hip/hip_runtime.h>

// ---------------------------------------------------------------------------
// Model constants (from reference): VOCAB=32000, CTX=1024, D=1024, H=16, L=4
// ---------------------------------------------------------------------------
#define VOCAB 32000
#define BATCH 2
#define SEQ   1024
#define DIM   1024
#define NHEAD 16
#define HD    64
#define NLAYER 4
#define MROWS (BATCH * SEQ)   // 2048 token rows

// ---------------------------------------------------------------------------
// WMMA vector types (gfx1250 wave32, CDNA5)
// ---------------------------------------------------------------------------
typedef __attribute__((ext_vector_type(16))) __bf16        v16bf;
typedef __attribute__((ext_vector_type(8)))  float         v8f;
typedef __attribute__((ext_vector_type(4)))  unsigned int  v4u;
typedef __attribute__((ext_vector_type(8)))  int           v8i;
typedef __attribute__((ext_vector_type(4)))  int           v4i;

union Frag16 {             // bf16 A/B fragment, filled from raw bf16 bits
    v16bf v;
    unsigned short u[16];
};

// float -> bf16 bits, round-to-nearest-even
__device__ __forceinline__ unsigned short f2bf(float f) {
    unsigned int u = __float_as_uint(f);
    u += 0x7FFFu + ((u >> 16) & 1u);
    return (unsigned short)(u >> 16);
}
__device__ __forceinline__ unsigned pack2bf(float a, float b) {
    return (unsigned)f2bf(a) | ((unsigned)f2bf(b) << 16);
}

// ISA 7.12.2: 16-bit A-matrix 16x32 — element i of a lane's v16bf maps to
// K = i + 8*((i>>3) + laneHalf).  Same map used for the 32x16 B operand
// (lane%16 selects the N column).
__device__ __forceinline__ int kmap(int i, int hlf) {
    return i + 8 * ((i >> 3) + hlf);
}

// ---------------------------------------------------------------------------
// Tensor Data Mover: 2D tile load  global(bf16, row stride = K elems)
//   -> LDS contiguous rows x 32 cols.  D# per CDNA5 ISA §8.
//   Issue from ONE wave only (EXEC ignored; one issue per executing wave),
//   completion via TENSORcnt.
// ---------------------------------------------------------------------------
#if __has_builtin(__builtin_amdgcn_tensor_load_to_lds)
#define HAVE_TDM 1
__device__ __forceinline__ void tdm_load_a(unsigned short* lds,
                                           const unsigned short* gsrc,
                                           int K, int rows) {
    unsigned long long ga = (unsigned long long)(uintptr_t)gsrc;
    unsigned lds_off = (unsigned)(uintptr_t)lds;   // low 32 bits = LDS byte addr
    v4u g0;
    g0[0] = 1u;                                    // count=1, user mode
    g0[1] = lds_off;                               // lds_addr [63:32]
    g0[2] = (unsigned)(ga & 0xFFFFFFFFu);          // global_addr [95:64]
    g0[3] = (unsigned)((ga >> 32) & 0x01FFFFFFu)   // global_addr [120:96]
          | (2u << 30);                            // type=2 ("image") [127:126]
    v8i g1;
    g1[0] = (1 << 16);                             // wg_mask=0, data_size=1 (2B)
    g1[1] = (int)(((unsigned)K & 0xFFFFu) << 16);  // tensor_dim0[15:0]
    g1[2] = (int)((((unsigned)K >> 16) & 0xFFFFu)  // tensor_dim0[31:16]
          | (((unsigned)MROWS & 0xFFFFu) << 16));  // tensor_dim1[15:0]
    g1[3] = (32 << 16);                            // tile_dim0 = 32 (K cols)
    g1[4] = rows & 0xFFFF;                         // tile_dim1 = rows, tile_dim2=0
    g1[5] = K;                                     // tensor_dim0_stride[31:0]
    g1[6] = 0;                                     // stride[47:32]=0, dim1_stride lo
    g1[7] = 0;
    v4i z4 = {};
#if defined(__clang_major__) && (__clang_major__ >= 23)
    v8i z8 = {};
    __builtin_amdgcn_tensor_load_to_lds(g0, g1, z4, z4, z8, 0);
#else
    __builtin_amdgcn_tensor_load_to_lds(g0, g1, z4, z4, 0);
#endif
}
#else
#define HAVE_TDM 0
#endif

// ---------------------------------------------------------------------------
// 1) Embedding + full-dim rotary:  h = rotary(tok_emb[x] + pos_emb)
//    grid = B*S blocks, 512 threads (one rotary pair each)
// ---------------------------------------------------------------------------
__global__ __launch_bounds__(512) void embed_rotary_kernel(
    const int* __restrict__ x, const float* __restrict__ tok_emb,
    const float* __restrict__ pos_emb, const float* __restrict__ inv_freq,
    const float* __restrict__ rbias, float* __restrict__ h)
{
    int row = blockIdx.x;               // b*S + s
    int s   = row % SEQ;
    int p   = threadIdx.x;              // pair index 0..511
    int tok = x[row];
    size_t tb = (size_t)tok * DIM;
    size_t pb = (size_t)s   * DIM;
    float a = tok_emb[tb + 2 * p]     + pos_emb[pb + 2 * p];
    float b = tok_emb[tb + 2 * p + 1] + pos_emb[pb + 2 * p + 1];
    float fr = (float)s * inv_freq[p] + rbias[(size_t)s * (DIM / 2) + p];
    float c = cosf(fr), sn = sinf(fr);
    size_t ob = (size_t)row * DIM;
    h[ob + 2 * p]     = a * c - b * sn;
    h[ob + 2 * p + 1] = a * sn + b * c;
}

// ---------------------------------------------------------------------------
// 2) RMS norm, f32 in -> bf16 out.  One block (256 thr) per token row.
//    Each thread: exactly one float4 load + one packed uint2 store.
// ---------------------------------------------------------------------------
__global__ __launch_bounds__(256) void rmsnorm_kernel(
    const float* __restrict__ in, const float* __restrict__ w,
    unsigned short* __restrict__ out)
{
    __shared__ float red[8];
    __shared__ float rs_sh;
    int row = blockIdx.x;
    int tid = threadIdx.x;
    const float4* xr = (const float4*)(in + (size_t)row * DIM);
    float4 xv = xr[tid];                                   // 256 * 4 = 1024
    float4 wv = ((const float4*)w)[tid];
    float s = xv.x * xv.x + xv.y * xv.y + xv.z * xv.z + xv.w * xv.w;
    #pragma unroll
    for (int m = 16; m >= 1; m >>= 1) s += __shfl_xor(s, m, 32);
    if ((tid & 31) == 0) red[tid >> 5] = s;
    __syncthreads();
    if (tid == 0) {
        float t = 0.f;
        #pragma unroll
        for (int i = 0; i < 8; ++i) t += red[i];
        rs_sh = rsqrtf(t / (float)DIM + 1e-8f);
    }
    __syncthreads();
    float rs = rs_sh;
    uint2 o;
    o.x = pack2bf(xv.x * rs * wv.x, xv.y * rs * wv.y);
    o.y = pack2bf(xv.z * rs * wv.z, xv.w * rs * wv.w);
    ((uint2*)(out + (size_t)row * DIM))[tid] = o;
}

// ---------------------------------------------------------------------------
// 3) WMMA-bf16 GEMM:  out[M,N] = A[M,K](bf16) * Bw[N,K]^T(f32->bf16)
//    + bias + res1 + res2, optional ReLU, f32 and/or bf16 outputs.
//    256 threads = 8 waves; block tile 128x128; K-step 32.
//    A tile staged by the Tensor Data Mover, B tile converted by VALU
//    (float4 b128 loads, packed uint2 LDS stores).
//    Each wave: 32x64 strip -> 8 WMMAs from 2 A-frags x 4 B-frags per K-step.
// ---------------------------------------------------------------------------
#define TMB 128
#define TNB 128
#define TKB 32

__global__ __launch_bounds__(256) void gemm_bf16_kernel(
    const unsigned short* __restrict__ A, const float* __restrict__ Bw,
    const float* __restrict__ bias, const float* __restrict__ res1,
    const float* __restrict__ res2, float* __restrict__ outF,
    unsigned short* __restrict__ outB, int M, int N, int K, int relu)
{
    __shared__ unsigned short lsA[TMB * TKB];   // 8 KB, TDM-filled
    __shared__ unsigned short lsB[TNB * TKB];   // 8 KB, VALU-converted
    int tid  = threadIdx.x;
    int wave = tid >> 5;
    int lane = tid & 31;
    int hlf  = lane >> 4;
    int lm   = lane & 15;
    int waveM = wave >> 1;                      // 0..3 -> 32-row strip
    int waveN = wave & 1;                       // 0..1 -> 64-col strip
    int m0 = blockIdx.y * TMB;
    int n0 = blockIdx.x * TNB;
    int rB = tid >> 3;                          // 0..31 (+32 per step)
    int cB = (tid & 7) * 4;                     // 0,4,...,28

    v8f acc[2][4] = {};

    for (int k0 = 0; k0 < K; k0 += TKB) {
#if HAVE_TDM
        if (wave == 0)                          // one TDM issue per block tile
            tdm_load_a(lsA, A + (size_t)m0 * K + k0, K, TMB);
#else
        #pragma unroll
        for (int it = 0; it < 4; ++it) {
            int r = rB + it * 32;
            *(uint4*)&lsA[r * TKB] = ((const uint4*)&A[(size_t)(m0 + r) * K + k0])[tid & 7 & 3];
        }
#endif
        // B: 128 rows x 32 f32 cols; each thread 4x(float4 load -> uint2 store)
        #pragma unroll
        for (int it = 0; it < 4; ++it) {
            int r = rB + it * 32;
            float4 w4 = *(const float4*)&Bw[(size_t)(n0 + r) * K + k0 + cB];
            uint2 pk;
            pk.x = pack2bf(w4.x, w4.y);
            pk.y = pack2bf(w4.z, w4.w);
            *(uint2*)&lsB[r * TKB + cB] = pk;
        }
        if (k0 + TKB < K)   // hint next weight tile toward L2 (global_prefetch)
            __builtin_prefetch(&Bw[(size_t)(n0 + (tid >> 1)) * K + k0 + TKB], 0, 1);
#if HAVE_TDM
        if (wave == 0) __builtin_amdgcn_s_wait_tensorcnt(0);
#endif
        __syncthreads();

        Frag16 fa[2];
        #pragma unroll
        for (int fm = 0; fm < 2; ++fm) {
            int ar = waveM * 32 + fm * 16 + lm;
            #pragma unroll
            for (int i = 0; i < 16; ++i) fa[fm].u[i] = lsA[ar * TKB + kmap(i, hlf)];
        }
        #pragma unroll
        for (int c4 = 0; c4 < 4; ++c4) {
            Frag16 fb;
            int br = waveN * 64 + c4 * 16 + lm;
            #pragma unroll
            for (int i = 0; i < 16; ++i) fb.u[i] = lsB[br * TKB + kmap(i, hlf)];
            #pragma unroll
            for (int fm = 0; fm < 2; ++fm)
                acc[fm][c4] = __builtin_amdgcn_wmma_f32_16x16x32_bf16(
                    false, fa[fm].v, false, fb.v, (short)0, acc[fm][c4], false, false);
        }
        __syncthreads();
    }

    // epilogue (C layout: lane -> col n=lm, vgpr r -> row m = r + 8*hlf)
    #pragma unroll
    for (int fm = 0; fm < 2; ++fm) {
        #pragma unroll
        for (int c4 = 0; c4 < 4; ++c4) {
            #pragma unroll
            for (int r = 0; r < 8; ++r) {
                int m = m0 + waveM * 32 + fm * 16 + r + 8 * hlf;
                int n = n0 + waveN * 64 + c4 * 16 + lm;
                float v = acc[fm][c4][r];
                if (bias) v += bias[n];
                size_t idx = (size_t)m * N + n;
                if (res1) v += res1[idx];
                if (res2) v += res2[idx];
                if (relu) v = fmaxf(v, 0.f);
                if (outF) outF[idx] = v;
                if (outB) outB[idx] = f2bf(v);
            }
        }
    }
}

// ---------------------------------------------------------------------------
// 4) Per-head rotary for q,k + transpose q/k/v to (B,H,S,HD) bf16 + zero mask
//    grid = B*S blocks, 512 threads (H*HD/2 pairs)
// ---------------------------------------------------------------------------
__global__ __launch_bounds__(512) void qkv_rotary_kernel(
    const float* __restrict__ q, const float* __restrict__ k,
    const float* __restrict__ v, const float* __restrict__ inv_freq,
    const float* __restrict__ rbias, const float* __restrict__ factor,
    int layer, unsigned short* __restrict__ qr, unsigned short* __restrict__ kr,
    unsigned short* __restrict__ vr, float* __restrict__ sz)
{
    int row = blockIdx.x;               // b*S + s
    int b = row / SEQ, s = row % SEQ;
    int p = threadIdx.x;
    int head = p >> 5;                  // HD/2 = 32 pairs per head
    int i    = p & 31;
    size_t ib = (size_t)row * DIM;
    int d = head * HD + 2 * i;
    float qa = q[ib + d], qb = q[ib + d + 1];
    float ka = k[ib + d], kb = k[ib + d + 1];
    float fr = (float)s * inv_freq[i] + rbias[(size_t)s * (HD / 2) + i];
    float c = cosf(fr), sn = sinf(fr);
    float qra = qa * c - qb * sn, qrb = qa * sn + qb * c;
    float kra = ka * c - kb * sn, krb = ka * sn + kb * c;
    size_t ob = (((size_t)b * NHEAD + head) * SEQ + s) * HD + 2 * i;
    qr[ob] = f2bf(qra); qr[ob + 1] = f2bf(qrb);
    kr[ob] = f2bf(kra); kr[ob + 1] = f2bf(krb);
    vr[ob] = f2bf(v[ib + d]); vr[ob + 1] = f2bf(v[ib + d + 1]);
    if (i == 0) {
        float f  = factor[layer];
        float zf = log1pf(expf(f));                     // softplus
        zf = fminf(fmaxf(zf, 1e-5f), 0.1f);             // clip
        sz[((size_t)b * NHEAD + head) * SEQ + s] = (kra == 0.0f) ? zf : 1.0f;
    }
}

// ---------------------------------------------------------------------------
// 5) Attention: one wave per (b, h, 16-query tile).  Flash-style streaming
//    over 32-key tiles with online softmax.  NOTE: reference masks
//    multiplicatively (masked logits become 0.0 and STAY in the softmax).
// ---------------------------------------------------------------------------
__global__ __launch_bounds__(32) void attn_kernel(
    const unsigned short* __restrict__ qr, const unsigned short* __restrict__ kr,
    const unsigned short* __restrict__ vr, const float* __restrict__ sz,
    unsigned short* __restrict__ attn_out)
{
    __shared__ unsigned short pls[16 * 32];   // P tile staged C-layout -> A-layout
    int lane = threadIdx.x;
    int hlf  = lane >> 4;
    int lm   = lane & 15;
    int q0 = blockIdx.x * 16;
    int hi = blockIdx.y;
    int bi = blockIdx.z;
    size_t hb = ((size_t)bi * NHEAD + hi) * SEQ;        // (B,H,S,HD) row base
    size_t zb = ((size_t)bi * NHEAD + hi) * SEQ;        // sz base

    // Q fragments (held for the whole key loop): K-dims [0,32) and [32,64)
    Frag16 fq0, fq1;
    #pragma unroll
    for (int i = 0; i < 16; ++i) {
        int dk = kmap(i, hlf);
        size_t qrow = (hb + q0 + lm) * HD;
        fq0.u[i] = qr[qrow + dk];
        fq1.u[i] = qr[qrow + 32 + dk];
    }

    v8f o[4] = {};
    float mrun[8], lrun[8];
    #pragma unroll
    for (int r = 0; r < 8; ++r) { mrun[r] = -3.0e38f; lrun[r] = 0.f; }

    for (int kb = 0; kb < SEQ; kb += 32) {
        v8f s0 = {}, s1 = {};
        {   // scores for keys [kb, kb+16) and [kb+16, kb+32)
            Frag16 fb;
            size_t k0r = (hb + kb + lm) * HD;
            size_t k1r = (hb + kb + 16 + lm) * HD;
            #pragma unroll
            for (int i = 0; i < 16; ++i) fb.u[i] = kr[k0r + kmap(i, hlf)];
            s0 = __builtin_amdgcn_wmma_f32_16x16x32_bf16(false, fq0.v, false, fb.v, (short)0, s0, false, false);
            #pragma unroll
            for (int i = 0; i < 16; ++i) fb.u[i] = kr[k0r + 32 + kmap(i, hlf)];
            s0 = __builtin_amdgcn_wmma_f32_16x16x32_bf16(false, fq1.v, false, fb.v, (short)0, s0, false, false);
            #pragma unroll
            for (int i = 0; i < 16; ++i) fb.u[i] = kr[k1r + kmap(i, hlf)];
            s1 = __builtin_amdgcn_wmma_f32_16x16x32_bf16(false, fq0.v, false, fb.v, (short)0, s1, false, false);
            #pragma unroll
            for (int i = 0; i < 16; ++i) fb.u[i] = kr[k1r + 32 + kmap(i, hlf)];
            s1 = __builtin_amdgcn_wmma_f32_16x16x32_bf16(false, fq1.v, false, fb.v, (short)0, s1, false, false);
        }

        float z0 = sz[zb + kb + lm];
        float z1 = sz[zb + kb + 16 + lm];
        float fr_[8];
        #pragma unroll
        for (int r = 0; r < 8; ++r) {
            int qrow = q0 + r + 8 * hlf;
            // scale = HD^-0.25 on q and k => 1/sqrt(HD) = 0.125 on scores
            float a0 = s0[r] * 0.125f * ((kb + lm)      <= qrow ? z0 : 0.f);
            float a1 = s1[r] * 0.125f * ((kb + 16 + lm) <= qrow ? z1 : 0.f);
            float mx = fmaxf(a0, a1);
            #pragma unroll
            for (int sh = 1; sh < 16; sh <<= 1) mx = fmaxf(mx, __shfl_xor(mx, sh, 32));
            float mn = fmaxf(mrun[r], mx);
            float fsc = __expf(mrun[r] - mn);
            float p0 = __expf(a0 - mn);
            float p1 = __expf(a1 - mn);
            s0[r] = p0; s1[r] = p1;
            float rs = p0 + p1;
            #pragma unroll
            for (int sh = 1; sh < 16; sh <<= 1) rs += __shfl_xor(rs, sh, 32);
            lrun[r] = lrun[r] * fsc + rs;
            mrun[r] = mn;
            fr_[r] = fsc;
        }
        #pragma unroll
        for (int c4 = 0; c4 < 4; ++c4)
            #pragma unroll
            for (int r = 0; r < 8; ++r) o[c4][r] *= fr_[r];

        // stage P (C layout) to LDS, re-read as A fragment
        #pragma unroll
        for (int r = 0; r < 8; ++r) {
            int m = r + 8 * hlf;
            pls[m * 32 + lm]      = f2bf(s0[r]);
            pls[m * 32 + 16 + lm] = f2bf(s1[r]);
        }
        __syncthreads();
        Frag16 fp;
        #pragma unroll
        for (int i = 0; i < 16; ++i) fp.u[i] = pls[lm * 32 + kmap(i, hlf)];
        #pragma unroll
        for (int c4 = 0; c4 < 4; ++c4) {
            Frag16 fv;
            #pragma unroll
            for (int i = 0; i < 16; ++i)
                fv.u[i] = vr[(hb + kb + kmap(i, hlf)) * HD + c4 * 16 + lm];
            o[c4] = __builtin_amdgcn_wmma_f32_16x16x32_bf16(
                false, fp.v, false, fv.v, (short)0, o[c4], false, false);
        }
        __syncthreads();
    }

    // normalize and write (B,S,D) bf16
    #pragma unroll
    for (int c4 = 0; c4 < 4; ++c4) {
        #pragma unroll
        for (int r = 0; r < 8; ++r) {
            int m = r + 8 * hlf;
            float val = o[c4][r] / lrun[r];
            attn_out[((size_t)(bi * SEQ + q0 + m)) * DIM + hi * HD + c4 * 16 + lm] = f2bf(val);
        }
    }
}

// ---------------------------------------------------------------------------
// Host-side orchestration
// ---------------------------------------------------------------------------
extern "C" void kernel_launch(void* const* d_in, const int* in_sizes, int n_in,
                              void* d_out, int out_size, void* d_ws, size_t ws_size,
                              hipStream_t stream) {
    (void)in_sizes; (void)n_in; (void)out_size; (void)ws_size;
    const int*   x         = (const int*)  d_in[0];
    const float* tok_emb   = (const float*)d_in[2];
    const float* pos_emb   = (const float*)d_in[3];
    const float* rot_inv   = (const float*)d_in[4];
    const float* rot_bias  = (const float*)d_in[5];
    const float* brot_inv  = (const float*)d_in[6];
    const float* brot_bias = (const float*)d_in[7];
    const float* lna_w     = (const float*)d_in[8];
    const float* q_w       = (const float*)d_in[9];
    const float* q_b       = (const float*)d_in[10];
    const float* k_w       = (const float*)d_in[11];
    const float* v_w       = (const float*)d_in[12];
    const float* v_b       = (const float*)d_in[13];
    const float* o_w       = (const float*)d_in[14];
    const float* o_b       = (const float*)d_in[15];
    const float* factor    = (const float*)d_in[16];
    const float* lnc_w     = (const float*)d_in[17];
    const float* mlp_w1    = (const float*)d_in[18];
    const float* mlp_b1    = (const float*)d_in[19];
    const float* mlp_w2    = (const float*)d_in[20];
    const float* mlp_b2    = (const float*)d_in[21];
    const float* ln_dec_w  = (const float*)d_in[22];

    // workspace carve-up
    char* ws = (char*)d_ws;
    auto carve = [&ws](size_t bytes) {
        char* p = ws;
        ws += (bytes + 255) & ~(size_t)255;
        return p;
    };
    const size_t MD  = (size_t)MROWS * DIM;
    float*          h     = (float*)         carve(MD * 4);
    float*          h1    = (float*)         carve(MD * 4);
    float*          qf    = (float*)         carve(MD * 4);
    float*          kf    = (float*)         carve(MD * 4);
    float*          vf    = (float*)         carve(MD * 4);
    unsigned short* hnorm = (unsigned short*)carve(MD * 2);
    unsigned short* qrb   = (unsigned short*)carve(MD * 2);
    unsigned short* krb   = (unsigned short*)carve(MD * 2);
    unsigned short* vrb   = (unsigned short*)carve(MD * 2);
    unsigned short* attnb = (unsigned short*)carve(MD * 2);
    unsigned short* mlp1b = (unsigned short*)carve(MD * 4 * 2);
    float*          szb   = (float*)         carve((size_t)BATCH * NHEAD * SEQ * 4);

    dim3 gemmBlk(256);
    dim3 gD (DIM / TNB,     MROWS / TMB);    // N=1024  (8  x 16)
    dim3 gF1(DIM * 4 / TNB, MROWS / TMB);    // N=4096  (32 x 16)
    dim3 gV (VOCAB / TNB,   MROWS / TMB);    // N=32000 (250 x 16)

    embed_rotary_kernel<<<dim3(MROWS), 512, 0, stream>>>(
        x, tok_emb, pos_emb, rot_inv, rot_bias, h);

    for (int l = 0; l < NLAYER; ++l) {
        const size_t wo = (size_t)l * DIM * DIM;
        // pre-attn RMS norm
        rmsnorm_kernel<<<dim3(MROWS), 256, 0, stream>>>(h, lna_w + (size_t)l * DIM, hnorm);
        // q, k, v projections
        gemm_bf16_kernel<<<gD, gemmBlk, 0, stream>>>(hnorm, q_w + wo, q_b + (size_t)l * DIM,
            nullptr, nullptr, qf, nullptr, MROWS, DIM, DIM, 0);
        gemm_bf16_kernel<<<gD, gemmBlk, 0, stream>>>(hnorm, k_w + wo, nullptr,
            nullptr, nullptr, kf, nullptr, MROWS, DIM, DIM, 0);
        gemm_bf16_kernel<<<gD, gemmBlk, 0, stream>>>(hnorm, v_w + wo, v_b + (size_t)l * DIM,
            nullptr, nullptr, vf, nullptr, MROWS, DIM, DIM, 0);
        // rotary + head transpose + zero-token mask
        qkv_rotary_kernel<<<dim3(MROWS), 512, 0, stream>>>(
            qf, kf, vf, brot_inv, brot_bias, factor, l, qrb, krb, vrb, szb);
        // attention
        attn_kernel<<<dim3(SEQ / 16, NHEAD, BATCH), 32, 0, stream>>>(
            qrb, krb, vrb, szb, attnb);
        // o-projection + residual (h1 = h + attn @ ow^T + ob)
        gemm_bf16_kernel<<<gD, gemmBlk, 0, stream>>>(attnb, o_w + wo, o_b + (size_t)l * DIM,
            h, nullptr, h1, nullptr, MROWS, DIM, DIM, 0);
        // MLP
        rmsnorm_kernel<<<dim3(MROWS), 256, 0, stream>>>(h1, lnc_w + (size_t)l * DIM, hnorm);
        gemm_bf16_kernel<<<gF1, gemmBlk, 0, stream>>>(hnorm,
            mlp_w1 + (size_t)l * 4 * DIM * DIM, mlp_b1 + (size_t)l * 4 * DIM,
            nullptr, nullptr, nullptr, mlp1b, MROWS, 4 * DIM, DIM, 1);
        // h = mlp2 + b2 + h1 + h_old   (in-place: each element read-then-written once)
        gemm_bf16_kernel<<<gD, gemmBlk, 0, stream>>>(mlp1b,
            mlp_w2 + (size_t)l * DIM * 4 * DIM, mlp_b2 + (size_t)l * DIM,
            h1, h, h, nullptr, MROWS, DIM, 4 * DIM, 0);
    }

    // final norm + logits
    rmsnorm_kernel<<<dim3(MROWS), 256, 0, stream>>>(h, ln_dec_w, hnorm);
    gemm_bf16_kernel<<<gV, gemmBlk, 0, stream>>>(hnorm, tok_emb, nullptr,
        nullptr, nullptr, (float*)d_out, nullptr, MROWS, VOCAB, DIM, 0);
}
